// G_SP_64974265254343
// MI455X (gfx1250) — matrix operand order repaired and checked by reading, not compiled, not run on hardware
//
#include <hip/hip_runtime.h>

// ---------------------------------------------------------------------------
// MI455X (gfx1250) implementation.
// All GEMMs use v_wmma_f32_16x16x32_bf16 (wave32, 16x16 tiles, K=32 step),
// with double-buffered LDS tiles staged via GLOBAL_LOAD_ASYNC_TO_LDS_B128
// (ASYNCcnt) when the toolchain exposes the builtin, sync copies otherwise.
// Pipeline:
//   1) cast dec/enc/Wv/W1/W2 to bf16 in workspace
//   2) v = enc @ Wv + bv                      (gemm_nn, EPI=1 -> bf16)
//   3) S = dec @ enc^T                        (scores_nt, f32 raw scores)
//   4) nonstandard masked softmax over M      (softmax_kernel -> bf16 attn)
//   5) g = dec * (1 + tanh(attn @ v))         (gemm_nn, EPI=3 -> bf16)
//   6) h = relu(g @ W1 + b1)                  (gemm_nn, EPI=2 -> bf16)
//   7) out = h @ W2 + b2                      (gemm_nn, EPI=0 -> f32 d_out)
// ---------------------------------------------------------------------------

#define BB 8
#define NN 2048
#define MM 2048
#define CC 512

typedef __attribute__((ext_vector_type(16))) __bf16 v16bf;
typedef __attribute__((ext_vector_type(8)))  float  v8f;
typedef int v4i_g __attribute__((vector_size(16)));   // matches builtin param type

#if defined(__gfx1250__) && __has_builtin(__builtin_amdgcn_global_load_async_to_lds_b128) && __has_builtin(__builtin_amdgcn_s_wait_asynccnt)
#define HAVE_ASYNC_LDS 1
#else
#define HAVE_ASYNC_LDS 0
#endif

union FragU {
    v16bf v;
    uint4 q[2];
    unsigned short us[16];
};

__device__ __forceinline__ unsigned short f2bf(float f) {
    unsigned u = __float_as_uint(f);
    unsigned r = u + 0x7FFFu + ((u >> 16) & 1u);   // round-to-nearest-even
    return (unsigned short)(r >> 16);
}

// 16-byte global -> LDS copy: async (ASYNCcnt) when available, sync otherwise.
__device__ __forceinline__ void cp16(unsigned short* ldst, const unsigned short* gsrc) {
#if HAVE_ASYNC_LDS
    __builtin_amdgcn_global_load_async_to_lds_b128(
        (__attribute__((address_space(1))) v4i_g*)gsrc,
        (__attribute__((address_space(3))) v4i_g*)ldst, 0, 0);
#else
    *(uint4*)ldst = *(const uint4*)gsrc;
#endif
}
__device__ __forceinline__ void cp32(unsigned short* ldst, const unsigned short* gsrc) {
    cp16(ldst, gsrc);
    cp16(ldst + 8, gsrc + 8);
}
__device__ __forceinline__ void async_wait() {
#if HAVE_ASYNC_LDS
    __builtin_amdgcn_s_wait_asynccnt(0);
#endif
}

// ---- fragment loads from LDS tiles ----------------------------------------
// lds_a layout: [128 rows][32 k] ushort (K contiguous per row)
// A 16x32 bf16 frag: lane<16 -> M=lane, elems = K 0..15
//                    lane>=16 -> M=lane-16, elems = K 16..31
__device__ __forceinline__ v16bf load_frag_a(const unsigned short* lds_a,
                                             int rowBase, int lane) {
    int r  = rowBase + (lane & 15);
    int kb = (lane < 16) ? 0 : 16;
    FragU f;
    f.q[0] = *(const uint4*)(lds_a + r * 32 + kb);        // K kb..kb+7
    f.q[1] = *(const uint4*)(lds_a + r * 32 + kb + 8);    // K kb+8..kb+15
    return f.v;
}

// lds_b layout: [128 n][32 k] ushort (K contiguous per output column)
// B 32x16 bf16 frag: lane<16 -> N=lane, elems = K 0..15
//                    lane>=16 -> N=lane-16, elems = K 16..31
__device__ __forceinline__ v16bf load_frag_b(const unsigned short* lds_b,
                                             int colBase, int lane) {
    int c  = colBase + (lane & 15);
    int kb = (lane < 16) ? 0 : 16;
    FragU f;
    const uint4* p = (const uint4*)(lds_b + c * 32 + kb);
    f.q[0] = p[0];
    f.q[1] = p[1];
    return f.v;
}

// ---- f32 -> bf16 cast (float4 vectorized; all sizes are multiples of 4) ----
__global__ __launch_bounds__(256) void f32_to_bf16_kernel(
    const float* __restrict__ in, unsigned short* __restrict__ out, int n4) {
    int i = blockIdx.x * 256 + threadIdx.x;
    if (i < n4) {
        float4 f = ((const float4*)in)[i];
        ushort4 o;
        o.x = f2bf(f.x); o.y = f2bf(f.y); o.z = f2bf(f.z); o.w = f2bf(f.w);
        ((ushort4*)out)[i] = o;
    }
}

// ---- scores: S[b] = dec[b] (NxC) x enc[b]^T (MxC)^T, f32 out ---------------
__global__ __launch_bounds__(256) void scores_nt_kernel(
    const unsigned short* __restrict__ Abf,   // dec bf16 [B,N,C]
    const unsigned short* __restrict__ Bbf,   // enc bf16 [B,M,C]
    float* __restrict__ S)                    // [B,N,M]
{
    __shared__ __align__(16) unsigned short lds_a[2][128 * 32];
    __shared__ __align__(16) unsigned short lds_b[2][128 * 32];

    const int b    = blockIdx.z;
    const int row0 = blockIdx.y * 128;
    const int col0 = blockIdx.x * 128;
    const unsigned short* A  = Abf + (size_t)b * NN * CC;
    const unsigned short* Bm = Bbf + (size_t)b * MM * CC;
    float* So = S + (size_t)b * NN * MM;

    const int tid  = threadIdx.x;
    const int lane = tid & 31;
    const int wave = tid >> 5;
    const int wr   = (wave >> 2) * 64;   // 0 / 64
    const int wc   = (wave & 3) * 32;    // 0 / 32 / 64 / 96

    const int lrow  = tid >> 1;          // 0..127
    const int lhalf = (tid & 1) * 16;    // 0 / 16 (ushorts)
    const unsigned short* gA = A  + (size_t)(row0 + lrow) * CC + lhalf;
    const unsigned short* gB = Bm + (size_t)(col0 + lrow) * CC + lhalf;
    unsigned short* sAoff = &lds_a[0][lrow * 32 + lhalf];
    unsigned short* sBoff = &lds_b[0][lrow * 32 + lhalf];
    const int bufStride = 128 * 32;

    v8f acc[4][2] = {};

    // prologue: stage k-block 0 into buffer 0
    cp32(sAoff, gA);
    cp32(sBoff, gB);
    async_wait();
    __syncthreads();

    const int nK = CC / 32;
    for (int kk = 0; kk < nK; ++kk) {
        const int cur = kk & 1;
        if (kk + 1 < nK) {     // async prefetch next k-block into other buffer
            const int nxt = cur ^ 1;
            cp32(sAoff + nxt * bufStride, gA + (kk + 1) * 32);
            cp32(sBoff + nxt * bufStride, gB + (kk + 1) * 32);
        }
        const unsigned short* la = lds_a[cur];
        const unsigned short* lb = lds_b[cur];
        v16bf bf0 = load_frag_b(lb, wc, lane);
        v16bf bf1 = load_frag_b(lb, wc + 16, lane);
#pragma unroll
        for (int i = 0; i < 4; ++i) {
            v16bf af = load_frag_a(la, wr + i * 16, lane);
            acc[i][0] = __builtin_amdgcn_wmma_f32_16x16x32_bf16(
                false, af, false, bf0, (short)0, acc[i][0], false, false);
            acc[i][1] = __builtin_amdgcn_wmma_f32_16x16x32_bf16(
                false, af, false, bf1, (short)0, acc[i][1], false, false);
        }
        async_wait();
        __syncthreads();
    }

    const int ln = lane & 15;
    const int mh = (lane < 16) ? 0 : 8;
#pragma unroll
    for (int i = 0; i < 4; ++i)
#pragma unroll
        for (int j = 0; j < 2; ++j) {
            int col = col0 + wc + j * 16 + ln;
#pragma unroll
            for (int e = 0; e < 8; ++e) {
                int row = row0 + wr + i * 16 + mh + e;
                So[(size_t)row * MM + col] = acc[i][j][e];
            }
        }
}

// ---- nonstandard masked softmax over last dim (M=2048) ---------------------
// a = s*mask ; softmax includes the zeros (exp(0) in denom) ; re-zero a==0.
__global__ __launch_bounds__(256) void softmax_kernel(
    const float* __restrict__ S, const float* __restrict__ mask,
    unsigned short* __restrict__ attnBf) {
    __shared__ float a[MM];
    __shared__ float red[256];
    const size_t row = blockIdx.x;
    const float* s = S + row * MM;
    const float* m = mask + row * MM;
    const int tid = threadIdx.x;

    float lmax = -3.4e38f;
#pragma unroll
    for (int i = tid; i < MM; i += 256) {
        float v = s[i] * m[i];
        a[i] = v;
        lmax = fmaxf(lmax, v);
    }
    red[tid] = lmax;
    __syncthreads();
    for (int off = 128; off > 0; off >>= 1) {
        if (tid < off) red[tid] = fmaxf(red[tid], red[tid + off]);
        __syncthreads();
    }
    const float rmax = red[0];
    __syncthreads();

    float lsum = 0.f;
#pragma unroll
    for (int i = tid; i < MM; i += 256) lsum += __expf(a[i] - rmax);
    red[tid] = lsum;
    __syncthreads();
    for (int off = 128; off > 0; off >>= 1) {
        if (tid < off) red[tid] += red[tid + off];
        __syncthreads();
    }
    const float inv = 1.f / red[0];

#pragma unroll
    for (int i = tid; i < MM; i += 256) {
        float v = a[i];
        float r = (v != 0.f) ? __expf(v - rmax) * inv : 0.f;
        attnBf[row * MM + i] = f2bf(r);
    }
}

// ---- generic NN GEMM: out[b] = A[b] (rows x K) x B[b] (K x 512) + epilogue -
// EPI 0: f32 out = acc + bias              (fc2 -> d_out)
// EPI 1: bf16 out = acc + bias             (v = enc@Wv+bv)
// EPI 2: bf16 out = relu(acc + bias)       (fc1)
// EPI 3: bf16 out = dec * (1 + tanh(acc))  (attn@v, gate + residual)
template <int EPI>
__global__ __launch_bounds__(256) void gemm_nn_kernel(
    const unsigned short* __restrict__ Abf, size_t strideA, int lda, int K,
    const unsigned short* __restrict__ Bbf, size_t strideB,
    const float* __restrict__ bias, const float* __restrict__ dec,
    unsigned short* __restrict__ outBf, float* __restrict__ outF, int rows) {
    __shared__ __align__(16) unsigned short lds_a[2][128 * 32];
    __shared__ __align__(16) unsigned short lds_b[2][128 * 32];

    const int batch = blockIdx.z;
    const int row0  = blockIdx.y * 128;
    const int col0  = blockIdx.x * 128;
    const unsigned short* A  = Abf + (size_t)batch * strideA;
    const unsigned short* Bw = Bbf + (size_t)batch * strideB;
    const size_t rowBase = (size_t)batch * rows;   // global output row base

    const int tid  = threadIdx.x;
    const int lane = tid & 31;
    const int wave = tid >> 5;
    const int wr   = (wave >> 2) * 64;
    const int wc   = (wave & 3) * 32;

    v8f acc[4][2] = {};

    const int lrow  = tid >> 1;
    const int lhalf = (tid & 1) * 16;
    const int bk    = tid >> 3;          // 0..31  (K index in tile)
    const int bch   = (tid & 7) * 16;    // col chunk base 0..112

    const unsigned short* gA = A + (size_t)(row0 + lrow) * lda + lhalf;
    const unsigned short* gB = Bw + (size_t)bk * CC + col0 + bch;
    unsigned short* sAoff = &lds_a[0][lrow * 32 + lhalf];
    const int bufStride = 128 * 32;

    // prologue: stage k-block 0 (A async, B transpose-scatter sync)
    cp32(sAoff, gA);
    {
        FragU t;
        t.q[0] = ((const uint4*)gB)[0];
        t.q[1] = ((const uint4*)gB)[1];
#pragma unroll
        for (int j = 0; j < 16; ++j)
            lds_b[0][(bch + j) * 32 + bk] = t.us[j];
    }
    async_wait();
    __syncthreads();

    const int nK = K / 32;
    for (int kk = 0; kk < nK; ++kk) {
        const int cur = kk & 1;
        const int nxt = cur ^ 1;
        FragU t;
        if (kk + 1 < nK) {
            // issue next B global loads + next A async-to-LDS before compute
            const unsigned short* gBn = gB + (size_t)(kk + 1) * 32 * CC;
            t.q[0] = ((const uint4*)gBn)[0];
            t.q[1] = ((const uint4*)gBn)[1];
            cp32(sAoff + nxt * bufStride, gA + (kk + 1) * 32);
        }
        const unsigned short* la = lds_a[cur];
        const unsigned short* lb = lds_b[cur];
        v16bf bf0 = load_frag_b(lb, wc, lane);
        v16bf bf1 = load_frag_b(lb, wc + 16, lane);
#pragma unroll
        for (int i = 0; i < 4; ++i) {
            v16bf af = load_frag_a(la, wr + i * 16, lane);
            acc[i][0] = __builtin_amdgcn_wmma_f32_16x16x32_bf16(
                false, af, false, bf0, (short)0, acc[i][0], false, false);
            acc[i][1] = __builtin_amdgcn_wmma_f32_16x16x32_bf16(
                false, af, false, bf1, (short)0, acc[i][1], false, false);
        }
        if (kk + 1 < nK) {   // scatter next B tile after compute (overlaps latency)
            unsigned short* lbn = lds_b[nxt];
#pragma unroll
            for (int j = 0; j < 16; ++j)
                lbn[(bch + j) * 32 + bk] = t.us[j];
        }
        async_wait();
        __syncthreads();
    }

    const int ln = lane & 15;
    const int mh = (lane < 16) ? 0 : 8;
#pragma unroll
    for (int i = 0; i < 4; ++i)
#pragma unroll
        for (int j = 0; j < 2; ++j) {
            const int col = col0 + wc + j * 16 + ln;
#pragma unroll
            for (int e = 0; e < 8; ++e) {
                const int row = row0 + wr + i * 16 + mh + e;
                const size_t o = (rowBase + row) * (size_t)CC + col;
                float x = acc[i][j][e];
                if (EPI == 0) {
                    outF[o] = x + bias[col];
                } else if (EPI == 1) {
                    outBf[o] = f2bf(x + bias[col]);
                } else if (EPI == 2) {
                    float y = x + bias[col];
                    outBf[o] = f2bf(y > 0.f ? y : 0.f);
                } else {
                    float d = dec[o];
                    outBf[o] = f2bf(d * (1.f + tanhf(x)));
                }
            }
        }
}

// ---------------------------------------------------------------------------
extern "C" void kernel_launch(void* const* d_in, const int* in_sizes, int n_in,
                              void* d_out, int out_size, void* d_ws, size_t ws_size,
                              hipStream_t stream) {
    const float* dec   = (const float*)d_in[0];   // [B,N,C]
    const float* enc   = (const float*)d_in[1];   // [B,M,C]
    const float* trans = (const float*)d_in[2];   // [B,N,M]
    const float* Wv    = (const float*)d_in[3];
    const float* bv    = (const float*)d_in[4];
    const float* W1    = (const float*)d_in[5];
    const float* b1    = (const float*)d_in[6];
    const float* W2    = (const float*)d_in[7];
    const float* b2    = (const float*)d_in[8];
    float* out = (float*)d_out;

    const size_t BNC = (size_t)BB * NN * CC;      // 8,388,608
    const size_t BMC = (size_t)BB * MM * CC;
    const size_t BNM = (size_t)BB * NN * MM;      // 33,554,432
    const size_t W   = (size_t)CC * CC;

    // workspace partition (256B aligned)
    char* base = (char*)d_ws;
    size_t o = 0;
    auto take = [&](size_t bytes) { char* p = base + o; o += (bytes + 255) & ~(size_t)255; return p; };
    unsigned short* dec_bf  = (unsigned short*)take(BNC * 2);
    unsigned short* enc_bf  = (unsigned short*)take(BMC * 2);
    unsigned short* wv_bf   = (unsigned short*)take(W * 2);
    unsigned short* w1_bf   = (unsigned short*)take(W * 2);
    unsigned short* w2_bf   = (unsigned short*)take(W * 2);
    unsigned short* v_bf    = (unsigned short*)take(BMC * 2);
    float*          Sf      = (float*)take(BNM * 4);
    unsigned short* attn_bf = (unsigned short*)take(BNM * 2);
    unsigned short* g_bf    = (unsigned short*)take(BNC * 2);
    unsigned short* h_bf    = (unsigned short*)take(BNC * 2);
    (void)ws_size; (void)n_in; (void)in_sizes; (void)out_size;

    // 1) casts (float4 path: counts are multiples of 4)
    f32_to_bf16_kernel<<<(int)((BNC / 4 + 255) / 256), 256, 0, stream>>>(dec, dec_bf, (int)(BNC / 4));
    f32_to_bf16_kernel<<<(int)((BMC / 4 + 255) / 256), 256, 0, stream>>>(enc, enc_bf, (int)(BMC / 4));
    f32_to_bf16_kernel<<<(int)((W / 4 + 255) / 256), 256, 0, stream>>>(Wv, wv_bf, (int)(W / 4));
    f32_to_bf16_kernel<<<(int)((W / 4 + 255) / 256), 256, 0, stream>>>(W1, w1_bf, (int)(W / 4));
    f32_to_bf16_kernel<<<(int)((W / 4 + 255) / 256), 256, 0, stream>>>(W2, w2_bf, (int)(W / 4));

    // 2) v = enc @ Wv + bv   (rows = B*M = 16384)
    gemm_nn_kernel<1><<<dim3(CC / 128, (BB * MM) / 128, 1), 256, 0, stream>>>(
        enc_bf, 0, CC, CC, wv_bf, 0, bv, nullptr, v_bf, nullptr, BB * MM);

    // 3) raw scores S = dec @ enc^T
    scores_nt_kernel<<<dim3(MM / 128, NN / 128, BB), 256, 0, stream>>>(dec_bf, enc_bf, Sf);

    // 4) masked softmax -> bf16 attn
    softmax_kernel<<<BB * NN, 256, 0, stream>>>(Sf, trans, attn_bf);

    // 5) g = dec * (1 + tanh(attn @ v))   (batched: rows=N, K=M)
    gemm_nn_kernel<3><<<dim3(CC / 128, NN / 128, BB), 256, 0, stream>>>(
        attn_bf, (size_t)NN * MM, MM, MM, v_bf, (size_t)MM * CC,
        nullptr, dec, g_bf, nullptr, NN);

    // 6) h = relu(g @ W1 + b1)
    gemm_nn_kernel<2><<<dim3(CC / 128, (BB * NN) / 128, 1), 256, 0, stream>>>(
        g_bf, 0, CC, CC, w1_bf, 0, b1, nullptr, h_bf, nullptr, BB * NN);

    // 7) out = h @ W2 + b2  (f32)
    gemm_nn_kernel<0><<<dim3(CC / 128, (BB * NN) / 128, 1), 256, 0, stream>>>(
        h_bf, 0, CC, CC, w2_bf, 0, b2, nullptr, nullptr, out, BB * NN);
}
